// LiftLocalSelfAttention_53730040873217
// MI455X (gfx1250) — matrix operand order repaired and checked by reading, not compiled, not run on hardware
//
#include <hip/hip_runtime.h>
#include <hip/hip_fp16.h>

typedef __attribute__((ext_vector_type(16))) _Float16 v16h;
typedef __attribute__((ext_vector_type(8)))  float    v8f;

#define B_   2
#define CIN_ 3
#define H_   48
#define W_   48
#define HW_  (H_*W_)          // 2304
#define MID_ 32
#define NH_  8
#define P_   7
#define PP_  49
#define OUTC_ 32
#define G_   4
#define C_   (MID_*NH_)       // 256
#define NPAD_ 208             // G*P*P=196 padded to 13*16
#define MROWS_ (B_*NH_*HW_)   // 36864 rows of the positional-score GEMM
#define MOUT_  (B_*G_*HW_)    // 18432 rows of the output-projection GEMM

#define TILE_  16             // pixel tile edge for the attention kernel
#define HALO_  (TILE_ + P_ - 1)          // 22
#define HPIX_  (HALO_ * HALO_)           // 484
#define SSTR_  34                        // f16 stride per halo pixel (17 dwords, coprime w/ 64 banks)

union U32H2 { unsigned int u; _Float16 h[2]; };

// ---------------------------------------------------------------------------
// Kernel 1: coordinate-embedding MLPs (1->16 -> LN -> swish -> 16->16) for the
// 4 rotations, packed as the f16 B-matrix [n'=g*49+k*7+l][c=0..31] (row half in
// c<16, col half in c>=16), plus wout_w f32->f16 conversion.
// ---------------------------------------------------------------------------
__device__ static inline void mlp16(float t,
                                    const float* __restrict__ w1, const float* __restrict__ b1,
                                    const float* __restrict__ gm, const float* __restrict__ bt,
                                    const float* __restrict__ w2, const float* __restrict__ b2,
                                    float* out16) {
  float h[16];
  float mu = 0.f;
  #pragma unroll
  for (int j = 0; j < 16; ++j) { h[j] = t * w1[j] + b1[j]; mu += h[j]; }
  mu *= (1.f / 16.f);
  float var = 0.f;
  #pragma unroll
  for (int j = 0; j < 16; ++j) { float d = h[j] - mu; var += d * d; }
  var *= (1.f / 16.f);
  float inv = rsqrtf(var + 1e-5f);
  #pragma unroll
  for (int j = 0; j < 16; ++j) {
    float v = (h[j] - mu) * inv * gm[j] + bt[j];
    h[j] = v * (1.f / (1.f + __expf(-v)));   // swish
  }
  #pragma unroll
  for (int d = 0; d < 16; ++d) {
    float acc = b2[d];
    #pragma unroll
    for (int j = 0; j < 16; ++j) acc += h[j] * w2[d * 16 + j];
    out16[d] = acc;
  }
}

__global__ void embed_prep_kernel(const float* __restrict__ rw1, const float* __restrict__ rb1,
                                  const float* __restrict__ rg,  const float* __restrict__ rbt,
                                  const float* __restrict__ rw2, const float* __restrict__ rb2,
                                  const float* __restrict__ cw1, const float* __restrict__ cb1,
                                  const float* __restrict__ cg,  const float* __restrict__ cbt,
                                  const float* __restrict__ cw2, const float* __restrict__ cb2,
                                  const float* __restrict__ wout_w,
                                  _Float16* __restrict__ emb, _Float16* __restrict__ wh) {
  int t = threadIdx.x;
  if (t < G_ * PP_) {
    int g = t / PP_, kl = t % PP_, k = kl / P_, l = kl % P_;
    float ik = (float)k * (1.f / 3.f) - 1.f;     // 2*k/(P-1)-1
    float il = (float)l * (1.f / 3.f) - 1.f;
    const float csA[4] = {1.f, 0.f, -1.f, 0.f};
    const float snA[4] = {0.f, 1.f, 0.f, -1.f};
    float rc = csA[g] * ik - snA[g] * il;
    float cc = snA[g] * ik + csA[g] * il;
    float ro[16], co[16];
    mlp16(rc, rw1, rb1, rg, rbt, rw2, rb2, ro);
    mlp16(cc, cw1, cb1, cg, cbt, cw2, cb2, co);
    _Float16* e = emb + (size_t)t * 32;          // column index n' == t
    #pragma unroll
    for (int c = 0; c < 16; ++c) { e[c] = (_Float16)ro[c]; e[16 + c] = (_Float16)co[c]; }
  } else if (t < NPAD_) {                        // zero-pad columns 196..207
    _Float16* e = emb + (size_t)t * 32;
    #pragma unroll
    for (int c = 0; c < 32; ++c) e[c] = (_Float16)0.f;
  }
  for (int idx = t; idx < OUTC_ * C_; idx += 256) wh[idx] = (_Float16)wout_w[idx];
}

// ---------------------------------------------------------------------------
// Kernel 2: 1x1 conv q/k/v, stored f16 in WMMA-A row layout [b][n][pixel][mid]
// (channel ch of the conv maps to mid = ch>>3, n = ch&7 per the reshape).
// ---------------------------------------------------------------------------
__global__ void qkv_kernel(const float* __restrict__ x,
                           const float* __restrict__ q_w, const float* __restrict__ q_b,
                           const float* __restrict__ k_w, const float* __restrict__ k_b,
                           const float* __restrict__ v_w, const float* __restrict__ v_b,
                           _Float16* __restrict__ qh, _Float16* __restrict__ kh,
                           _Float16* __restrict__ vh) {
  int bp = blockIdx.x * blockDim.x + threadIdx.x;   // 0..B*HW-1
  int b = bp / HW_, p = bp % HW_;
  float x0 = x[(b * 3 + 0) * HW_ + p];
  float x1 = x[(b * 3 + 1) * HW_ + p];
  float x2 = x[(b * 3 + 2) * HW_ + p];
  for (int ch = 0; ch < C_; ++ch) {
    int mid = ch >> 3, n = ch & 7;
    size_t idx = ((size_t)((b * NH_ + n) * HW_ + p)) * MID_ + mid;
    float qv = q_w[ch * 3] * x0 + q_w[ch * 3 + 1] * x1 + q_w[ch * 3 + 2] * x2 + q_b[ch];
    float kv = k_w[ch * 3] * x0 + k_w[ch * 3 + 1] * x1 + k_w[ch * 3 + 2] * x2 + k_b[ch];
    float vv = v_w[ch * 3] * x0 + v_w[ch * 3 + 1] * x1 + v_w[ch * 3 + 2] * x2 + v_b[ch];
    qh[idx] = (_Float16)qv;
    kh[idx] = (_Float16)kv;
    vh[idx] = (_Float16)vv;
  }
}

// ---------------------------------------------------------------------------
// Kernel 3: positional-score GEMM  pos[36864 x 208] = Q[36864 x 32] * E[32 x 208]
// v_wmma_f32_16x16x32_f16, one wave per 16-row strip, 13 N-tiles, single K-step.
// ---------------------------------------------------------------------------
__global__ void pos_gemm_kernel(const _Float16* __restrict__ qh,
                                const _Float16* __restrict__ emb,
                                float* __restrict__ pos) {
  int wave = (blockIdx.x * blockDim.x + threadIdx.x) >> 5;
  int lane = threadIdx.x & 31;
  int m0 = wave * 16;
  int lrow = lane & 15;
  bool hi = lane >= 16;

  union { v16h h; unsigned int u[8]; } a, bf;
  const _Float16* arow = qh + (size_t)(m0 + lrow) * 32;
  #pragma unroll
  for (int r = 0; r < 8; ++r) {                 // A 16x32 f16 lane layout
    int K0 = 2 * r + (hi ? 16 : 0);
    a.u[r] = *(const unsigned int*)(arow + K0);
  }
  for (int nt = 0; nt < 13; ++nt) {
    const _Float16* bcol = emb + (size_t)(nt * 16 + lrow) * 32;
    #pragma unroll
    for (int r = 0; r < 8; ++r) {               // B 32x16 f16 lane layout
      int K0 = 2 * r + (hi ? 16 : 0);
      bf.u[r] = *(const unsigned int*)(bcol + K0);
    }
    v8f c = {};
    c = __builtin_amdgcn_wmma_f32_16x16x32_f16(false, a.h, false, bf.h,
                                               (short)0, c, false, false);
    float* prow = pos + (size_t)m0 * NPAD_ + nt * 16 + lrow;
    #pragma unroll
    for (int r = 0; r < 8; ++r) {               // C/D f32 lane layout
      int Mr = r + (hi ? 8 : 0);
      prow[(size_t)Mr * NPAD_] = c[r];
    }
  }
}

// ---------------------------------------------------------------------------
// Kernel 4: fused content scores + mask + softmax + value aggregation,
// halo-tiled through LDS. One block = one (b, head) x 16x16 pixel tile.
// K/V halos (22x22 pixels x 32 f16, padded to 34 f16 -> 17-dword lane stride,
// conflict-free) staged once in LDS (~66 KB of the 320 KB WGP pool), then each
// thread handles one pixel for all 4 rotations (content computed once).
// Writes v_out f16 in A-layout [row = (b*G+g)*HW + p][c = mid*8 + n].
// ---------------------------------------------------------------------------
__global__ void attn_kernel(const _Float16* __restrict__ qh, const _Float16* __restrict__ kh,
                            const _Float16* __restrict__ vh, const float* __restrict__ pos,
                            _Float16* __restrict__ vo) {
  __shared__ _Float16 sK[HPIX_ * SSTR_];
  __shared__ _Float16 sV[HPIX_ * SSTR_];

  int tid = threadIdx.x;
  int tile = blockIdx.x % 9;                 // 3x3 tiles of 16x16 over 48x48
  int bn   = blockIdx.x / 9;                 // (b*NH + n)
  int i0 = (tile / 3) * TILE_, j0 = (tile % 3) * TILE_;
  size_t bnrow = (size_t)bn * HW_;
  int b = bn / NH_, n = bn % NH_;

  // ---- stage K/V halo into LDS (zeros outside the image) ----
  for (int idx = tid; idx < HPIX_; idx += 256) {
    int ii = i0 - 3 + idx / HALO_;
    int jj = j0 - 3 + idx % HALO_;
    _Float16* dk = sK + idx * SSTR_;
    _Float16* dv = sV + idx * SSTR_;
    if (ii >= 0 && ii < H_ && jj >= 0 && jj < W_) {
      const unsigned int* sk = (const unsigned int*)(kh + (bnrow + ii * W_ + jj) * MID_);
      const unsigned int* sv = (const unsigned int*)(vh + (bnrow + ii * W_ + jj) * MID_);
      #pragma unroll
      for (int r = 0; r < 16; ++r) {
        *(unsigned int*)(dk + 2 * r) = sk[r];
        *(unsigned int*)(dv + 2 * r) = sv[r];
      }
    } else {
      #pragma unroll
      for (int r = 0; r < 16; ++r) {
        *(unsigned int*)(dk + 2 * r) = 0u;
        *(unsigned int*)(dv + 2 * r) = 0u;
      }
    }
  }
  __syncthreads();

  int ty = tid / TILE_, tx = tid % TILE_;
  int i = i0 + ty, j = j0 + tx;
  int p = i * W_ + j;

  // ---- content scores (shared by all 4 rotations) ----
  float qf[32];
  {
    const _Float16* qp = qh + (bnrow + p) * MID_;
    #pragma unroll
    for (int c2 = 0; c2 < 16; ++c2) {
      U32H2 u; u.u = *(const unsigned int*)(qp + 2 * c2);
      qf[2 * c2] = (float)u.h[0]; qf[2 * c2 + 1] = (float)u.h[1];
    }
  }
  float content[PP_];
  for (int kl = 0; kl < PP_; ++kl) {
    int dk = kl / P_, dl = kl % P_;
    const _Float16* kp = sK + ((ty + dk) * HALO_ + tx + dl) * SSTR_;
    float acc = 0.f;
    #pragma unroll
    for (int c2 = 0; c2 < 16; ++c2) {
      U32H2 u; u.u = *(const unsigned int*)(kp + 2 * c2);
      acc += qf[2 * c2] * (float)u.h[0] + qf[2 * c2 + 1] * (float)u.h[1];
    }
    content[kl] = acc;
  }

  const float* prow0 = pos + (bnrow + p) * NPAD_;

  for (int g = 0; g < G_; ++g) {
    const float* prow = prow0 + g * PP_;
    float s[PP_];
    float mx = -3.4e38f;
    for (int kl = 0; kl < PP_; ++kl) {
      int ii = i + kl / P_ - 3, jj = j + kl % P_ - 3;
      float sc = -1e30f;
      if (ii >= 0 && ii < H_ && jj >= 0 && jj < W_)
        sc = (content[kl] + prow[kl]) * 0.5773502691896258f;   // 1/sqrt(CIN=3)
      s[kl] = sc;
      mx = fmaxf(mx, sc);
    }
    float den = 0.f;
    for (int kl = 0; kl < PP_; ++kl) { float e = __expf(s[kl] - mx); s[kl] = e; den += e; }
    float inv = 1.f / den;

    float acc[32];
    #pragma unroll
    for (int c = 0; c < 32; ++c) acc[c] = 0.f;
    for (int kl = 0; kl < PP_; ++kl) {
      float w = s[kl] * inv;                     // invalid positions: w == 0, LDS holds zeros
      int dk = kl / P_, dl = kl % P_;
      const _Float16* vp = sV + ((ty + dk) * HALO_ + tx + dl) * SSTR_;
      #pragma unroll
      for (int c2 = 0; c2 < 16; ++c2) {
        U32H2 u; u.u = *(const unsigned int*)(vp + 2 * c2);
        acc[2 * c2]     += w * (float)u.h[0];
        acc[2 * c2 + 1] += w * (float)u.h[1];
      }
    }
    _Float16* op = vo + ((size_t)((b * G_ + g) * HW_) + p) * C_ + n;
    #pragma unroll
    for (int mid = 0; mid < 32; ++mid) op[mid * NH_] = (_Float16)acc[mid];
  }
}

// ---------------------------------------------------------------------------
// Kernel 5: output projection GEMM  out[18432 x 32] = Vout[18432 x 256] * Wout^T
// v_wmma_f32_16x16x32_f16, 8 chained K-steps, 2 N-tiles, + bias, scatter-store
// into [B, OUT, G, H, W].
// ---------------------------------------------------------------------------
__global__ void wout_gemm_kernel(const _Float16* __restrict__ vo,
                                 const _Float16* __restrict__ wh,
                                 const float* __restrict__ wout_b,
                                 float* __restrict__ out) {
  int wave = (blockIdx.x * blockDim.x + threadIdx.x) >> 5;
  int lane = threadIdx.x & 31;
  int m0 = wave * 16;
  int lrow = lane & 15;
  bool hi = lane >= 16;

  v8f c0 = {}, c1 = {};
  const _Float16* arow = vo + (size_t)(m0 + lrow) * C_;
  for (int kt = 0; kt < 8; ++kt) {
    union { v16h h; unsigned int u[8]; } a, b0, b1;
    #pragma unroll
    for (int r = 0; r < 8; ++r) {
      int K0 = kt * 32 + 2 * r + (hi ? 16 : 0);
      a.u[r] = *(const unsigned int*)(arow + K0);
    }
    int Kb = kt * 32 + (hi ? 16 : 0);
    const _Float16* b0p = wh + (size_t)lrow * C_ + Kb;
    const _Float16* b1p = wh + (size_t)(16 + lrow) * C_ + Kb;
    #pragma unroll
    for (int r = 0; r < 8; ++r) {
      b0.u[r] = *(const unsigned int*)(b0p + 2 * r);
      b1.u[r] = *(const unsigned int*)(b1p + 2 * r);
    }
    c0 = __builtin_amdgcn_wmma_f32_16x16x32_f16(false, a.h, false, b0.h,
                                                (short)0, c0, false, false);
    c1 = __builtin_amdgcn_wmma_f32_16x16x32_f16(false, a.h, false, b1.h,
                                                (short)0, c1, false, false);
  }
  int o0 = lrow, o1 = 16 + lrow;
  float bb0 = wout_b[o0], bb1 = wout_b[o1];
  #pragma unroll
  for (int r = 0; r < 8; ++r) {
    int row = m0 + r + (hi ? 8 : 0);              // row = (b*G + g)*HW + p
    int bg = row / HW_, p = row % HW_;
    int bb = bg / G_, g = bg % G_;
    size_t base = ((size_t)bb * OUTC_ * G_ + g) * HW_ + p;
    out[base + (size_t)o0 * G_ * HW_] = c0[r] + bb0;
    out[base + (size_t)o1 * G_ * HW_] = c1[r] + bb1;
  }
}

// ---------------------------------------------------------------------------
extern "C" void kernel_launch(void* const* d_in, const int* in_sizes, int n_in,
                              void* d_out, int out_size, void* d_ws, size_t ws_size,
                              hipStream_t stream) {
  const float* x      = (const float*)d_in[0];
  const float* q_w    = (const float*)d_in[1];
  const float* q_b    = (const float*)d_in[2];
  const float* k_w    = (const float*)d_in[3];
  const float* k_b    = (const float*)d_in[4];
  const float* v_w    = (const float*)d_in[5];
  const float* v_b    = (const float*)d_in[6];
  const float* wout_w = (const float*)d_in[7];
  const float* wout_b = (const float*)d_in[8];
  const float* rw1 = (const float*)d_in[9];
  const float* rb1 = (const float*)d_in[10];
  const float* rg  = (const float*)d_in[11];
  const float* rbt = (const float*)d_in[12];
  const float* rw2 = (const float*)d_in[13];
  const float* rb2 = (const float*)d_in[14];
  const float* cw1 = (const float*)d_in[15];
  const float* cb1 = (const float*)d_in[16];
  const float* cg  = (const float*)d_in[17];
  const float* cbt = (const float*)d_in[18];
  const float* cw2 = (const float*)d_in[19];
  const float* cb2 = (const float*)d_in[20];
  float* out = (float*)d_out;

  // Workspace carving (all freshly written every call; ~47 MB total).
  char* ws = (char*)d_ws;
  size_t o_qh  = 0;
  size_t o_kh  = o_qh  + (size_t)MROWS_ * MID_ * sizeof(_Float16);
  size_t o_vh  = o_kh  + (size_t)MROWS_ * MID_ * sizeof(_Float16);
  size_t o_emb = o_vh  + (size_t)MROWS_ * MID_ * sizeof(_Float16);
  size_t o_wh  = o_emb + (size_t)NPAD_ * MID_ * sizeof(_Float16);
  size_t o_pos = o_wh  + (size_t)OUTC_ * C_ * sizeof(_Float16);
  size_t o_vo  = o_pos + (size_t)MROWS_ * NPAD_ * sizeof(float);

  _Float16* qh  = (_Float16*)(ws + o_qh);
  _Float16* kh  = (_Float16*)(ws + o_kh);
  _Float16* vh  = (_Float16*)(ws + o_vh);
  _Float16* emb = (_Float16*)(ws + o_emb);
  _Float16* wh  = (_Float16*)(ws + o_wh);
  float*    pos = (float*)(ws + o_pos);
  _Float16* vo  = (_Float16*)(ws + o_vo);

  embed_prep_kernel<<<1, 256, 0, stream>>>(rw1, rb1, rg, rbt, rw2, rb2,
                                           cw1, cb1, cg, cbt, cw2, cb2,
                                           wout_w, emb, wh);
  qkv_kernel<<<(B_ * HW_) / 256, 256, 0, stream>>>(x, q_w, q_b, k_w, k_b, v_w, v_b,
                                                   qh, kh, vh);
  pos_gemm_kernel<<<(MROWS_ / 16) / 8, 256, 0, stream>>>(qh, emb, pos);       // 288 blocks
  attn_kernel<<<B_ * NH_ * 9, 256, 0, stream>>>(qh, kh, vh, pos, vo);         // 144 blocks
  wout_gemm_kernel<<<(MOUT_ / 16) / 8, 256, 0, stream>>>(vo, wh, wout_b, out);// 144 blocks
}
